// LSTMModel_46986942218775
// MI455X (gfx1250) — matrix operand order repaired and checked by reading, not compiled, not run on hardware
//
#include <hip/hip_runtime.h>

typedef __attribute__((ext_vector_type(16))) _Float16 v16h;
typedef __attribute__((ext_vector_type(8)))  float    v8f;

#define B_TOT   131072
#define T_IN    24
#define F_IN    6
#define H       32
#define SEQ_OUT 3

#define NW   8            // waves per block
#define NT   (NW * 32)    // threads per block
#define BPB  (NW * 16)    // batch rows per block (16 per wave = WMMA M)
#define HS   33           // padded row stride (floats) for per-wave h buffer

__device__ __forceinline__ float sigm(float x) {
    float e = __expf(-x);
    return __builtin_amdgcn_rcpf(1.f + e);
}
__device__ __forceinline__ float tanh_fast(float x) {
    float e = __expf(-2.f * x);
    return 2.f * __builtin_amdgcn_rcpf(1.f + e) - 1.f;
}

// Stage W (rows = 128 gate outputs, cols = in_feat) as transposed, zero-padded
// f16 B-fragments in WMMA lane order: dst[(t*32 + lane)*16 + e] = W[N][K]
// with N = t*16 + (lane&15), K = (lane>>4)*16 + e  (K >= in_feat -> 0).
__device__ __forceinline__ void stage_weight(const float* __restrict__ W,
                                             int in_feat, _Float16* dst) {
    for (int idx = threadIdx.x; idx < 8 * 32; idx += NT) {
        int t = idx >> 5, l = idx & 31;
        int N = t * 16 + (l & 15);
        int kbase = (l >> 4) * 16;
        _Float16* p = dst + idx * 16;
#pragma unroll
        for (int e = 0; e < 16; ++e) {
            int K = kbase + e;
            float v = (K < in_feat) ? W[N * in_feat + K] : 0.f;
            p[e] = (_Float16)v;
        }
    }
}

__device__ __forceinline__ v16h load_frag(const _Float16* base, int t, int lane) {
    return *(const v16h*)(base + (t * 32 + lane) * 16);
}

// One column-half of an LSTM cell step: gate tiles {cs, 2+cs, 4+cs, 6+cs}
// covering output columns cs*16 + (0..15). Updates cell state `c` (8 rows per
// lane, D layout) and writes new h values into the per-wave LDS buffer.
__device__ __forceinline__ void lstm_half(const v16h Ax, const v16h Ah, v8f& c,
                                          int cs,
                                          const _Float16* Bx,
                                          const _Float16* Bh,
                                          float bi, float bf, float bg, float bo,
                                          int lane, float* hl) {
    const int ln = lane & 15;
    const int hf = lane >> 4;
    const v8f z = {};

    v8f gi, gf, gg, go;
    {
        v16h bx = load_frag(Bx, 0 + cs, lane);
        gi = __builtin_amdgcn_wmma_f32_16x16x32_f16(false, Ax, false, bx,
                                                    (short)0, z, false, false);
        v16h bh = load_frag(Bh, 0 + cs, lane);
        gi = __builtin_amdgcn_wmma_f32_16x16x32_f16(false, Ah, false, bh,
                                                    (short)0, gi, false, false);
    }
    {
        v16h bx = load_frag(Bx, 2 + cs, lane);
        gf = __builtin_amdgcn_wmma_f32_16x16x32_f16(false, Ax, false, bx,
                                                    (short)0, z, false, false);
        v16h bh = load_frag(Bh, 2 + cs, lane);
        gf = __builtin_amdgcn_wmma_f32_16x16x32_f16(false, Ah, false, bh,
                                                    (short)0, gf, false, false);
    }
    {
        v16h bx = load_frag(Bx, 4 + cs, lane);
        gg = __builtin_amdgcn_wmma_f32_16x16x32_f16(false, Ax, false, bx,
                                                    (short)0, z, false, false);
        v16h bh = load_frag(Bh, 4 + cs, lane);
        gg = __builtin_amdgcn_wmma_f32_16x16x32_f16(false, Ah, false, bh,
                                                    (short)0, gg, false, false);
    }
    {
        v16h bx = load_frag(Bx, 6 + cs, lane);
        go = __builtin_amdgcn_wmma_f32_16x16x32_f16(false, Ax, false, bx,
                                                    (short)0, z, false, false);
        v16h bh = load_frag(Bh, 6 + cs, lane);
        go = __builtin_amdgcn_wmma_f32_16x16x32_f16(false, Ah, false, bh,
                                                    (short)0, go, false, false);
    }

#pragma unroll
    for (int r = 0; r < 8; ++r) {
        float ig = sigm(gi[r] + bi);
        float fg = sigm(gf[r] + bf);
        float g2 = tanh_fast(gg[r] + bg);
        float og = sigm(go[r] + bo);
        float cn = fg * c[r] + ig * g2;
        c[r] = cn;
        hl[(r + hf * 8) * HS + cs * 16 + ln] = og * tanh_fast(cn);
    }
}

// Full LSTM cell step; bs = 8 per-lane bias scalars ordered
// [i_A, f_A, g_A, o_A, i_B, f_B, g_B, o_B].
__device__ __forceinline__ void lstm_step(const v16h Ax, v16h& Ah,
                                          v8f& cA, v8f& cB,
                                          const _Float16* Bx,
                                          const _Float16* Bh,
                                          const float* bs,
                                          int lane, float* hl) {
    const int ln = lane & 15;
    const int hf = lane >> 4;

    lstm_half(Ax, Ah, cA, 0, Bx, Bh, bs[0], bs[1], bs[2], bs[3], lane, hl);
    __builtin_amdgcn_sched_barrier(0);
    lstm_half(Ax, Ah, cB, 1, Bx, Bh, bs[4], bs[5], bs[6], bs[7], lane, hl);
    __builtin_amdgcn_sched_barrier(0);

    // Re-gather new h into WMMA A layout (16-bit A 16x32 mapping).
    v16h na;
#pragma unroll
    for (int e = 0; e < 16; ++e) {
        int K = hf * 16 + e;
        na[e] = (_Float16)hl[ln * HS + K];
    }
    Ah = na;
}

__global__ __launch_bounds__(NT) void lstm_enc_dec(
    const float* __restrict__ x,
    const float* eW0, const float* eU0, const float* ebi0, const float* ebh0,
    const float* eW1, const float* eU1, const float* ebi1, const float* ebh1,
    const float* dW0, const float* dU0, const float* dbi0, const float* dbh0,
    const float* dW1, const float* dU1, const float* dbi1, const float* dbh1,
    const float* fcW, const float* fcB,
    float* __restrict__ out) {
    __shared__ __align__(32) _Float16 wfrag[4][8 * 32 * 16];  // 32 KB
    __shared__ float biasv[2][128];
    __shared__ float hbuf[NW][16 * HS];
    __shared__ float fwlds[H];

    const int lane = threadIdx.x & 31;
    const int wv   = threadIdx.x >> 5;
    const int ln   = lane & 15;
    const int hf   = lane >> 4;
    const int b0   = blockIdx.x * BPB + wv * 16;
    float* hl = &hbuf[wv][0];

    // ---- stage encoder weights / biases ----
    stage_weight(eW0, F_IN, wfrag[0]);
    stage_weight(eU0, H,    wfrag[1]);
    stage_weight(eW1, H,    wfrag[2]);
    stage_weight(eU1, H,    wfrag[3]);
    for (int n = threadIdx.x; n < 128; n += NT) {
        biasv[0][n] = ebi0[n] + ebh0[n];
        biasv[1][n] = ebi1[n] + ebh1[n];
    }
    if (threadIdx.x < H) fwlds[threadIdx.x] = fcW[threadIdx.x];
    __syncthreads();

    // Per-lane bias scalars: bias for gate-tile (2*gate + cs) at column ln.
    float bs0[8], bs1[8];
#pragma unroll
    for (int g = 0; g < 4; ++g) {
        bs0[g]     = biasv[0][(2 * g + 0) * 16 + ln];
        bs0[4 + g] = biasv[0][(2 * g + 1) * 16 + ln];
        bs1[g]     = biasv[1][(2 * g + 0) * 16 + ln];
        bs1[4 + g] = biasv[1][(2 * g + 1) * 16 + ln];
    }

    v8f  cA0 = {}, cB0 = {}, cA1 = {}, cB1 = {};
    v16h Ah0 = {}, Ah1 = {};

    const float* xrow = x + (size_t)(b0 + ln) * (T_IN * F_IN);

    // ---- encoder ----
#pragma unroll 1
    for (int t = 0; t < T_IN; ++t) {
        v16h Ax = {};  // A layout: lanes 0-15 carry K=0..7 in e=0..7; K<6 valid
#pragma unroll
        for (int e = 0; e < F_IN; ++e) {
            float v = xrow[t * F_IN + e];
            Ax[e] = hf ? (_Float16)0.f : (_Float16)v;
        }
        lstm_step(Ax,  Ah0, cA0, cB0, wfrag[0], wfrag[1], bs0, lane, hl);
        __builtin_amdgcn_sched_barrier(0);
        lstm_step(Ah0, Ah1, cA1, cB1, wfrag[2], wfrag[3], bs1, lane, hl);
        __builtin_amdgcn_sched_barrier(0);
    }

    // ---- swap in decoder weights / biases ----
    __syncthreads();
    stage_weight(dW0, 1, wfrag[0]);
    stage_weight(dU0, H, wfrag[1]);
    stage_weight(dW1, H, wfrag[2]);
    stage_weight(dU1, H, wfrag[3]);
    for (int n = threadIdx.x; n < 128; n += NT) {
        biasv[0][n] = dbi0[n] + dbh0[n];
        biasv[1][n] = dbi1[n] + dbh1[n];
    }
    __syncthreads();

#pragma unroll
    for (int g = 0; g < 4; ++g) {
        bs0[g]     = biasv[0][(2 * g + 0) * 16 + ln];
        bs0[4 + g] = biasv[0][(2 * g + 1) * 16 + ln];
        bs1[g]     = biasv[1][(2 * g + 0) * 16 + ln];
        bs1[4 + g] = biasv[1][(2 * g + 1) * 16 + ln];
    }

    // ---- decoder ----
    float fcb    = fcB[0];
    float dec_in = xrow[(T_IN - 1) * F_IN + 0];  // x[:, -1, 0]

#pragma unroll 1
    for (int s = 0; s < SEQ_OUT; ++s) {
        v16h Ax = {};
        Ax[0] = hf ? (_Float16)0.f : (_Float16)dec_in;
        lstm_step(Ax,  Ah0, cA0, cB0, wfrag[0], wfrag[1], bs0, lane, hl);
        __builtin_amdgcn_sched_barrier(0);
        lstm_step(Ah0, Ah1, cA1, cB1, wfrag[2], wfrag[3], bs1, lane, hl);
        __builtin_amdgcn_sched_barrier(0);

        // FC: pred[row] = sum_k h1[row][k] * fcW[k] + fcb.
        // h1 is in hl; lanes ln and ln+16 each reduce 16 columns of row ln.
        float part = 0.f;
#pragma unroll
        for (int j = 0; j < 16; ++j)
            part += hl[ln * HS + hf * 16 + j] * fwlds[hf * 16 + j];
        float other = __shfl_xor(part, 16);
        float pred = part + other + fcb;

        if (hf == 0) out[(size_t)(b0 + ln) * SEQ_OUT + s] = pred;
        dec_in = pred;  // every lane holds pred for its row ln
    }
}

extern "C" void kernel_launch(void* const* d_in, const int* in_sizes, int n_in,
                              void* d_out, int out_size, void* d_ws, size_t ws_size,
                              hipStream_t stream) {
    const float* x    = (const float*)d_in[0];
    const float* eW0  = (const float*)d_in[1];
    const float* eU0  = (const float*)d_in[2];
    const float* ebi0 = (const float*)d_in[3];
    const float* ebh0 = (const float*)d_in[4];
    const float* eW1  = (const float*)d_in[5];
    const float* eU1  = (const float*)d_in[6];
    const float* ebi1 = (const float*)d_in[7];
    const float* ebh1 = (const float*)d_in[8];
    const float* dW0  = (const float*)d_in[9];
    const float* dU0  = (const float*)d_in[10];
    const float* dbi0 = (const float*)d_in[11];
    const float* dbh0 = (const float*)d_in[12];
    const float* dW1  = (const float*)d_in[13];
    const float* dU1  = (const float*)d_in[14];
    const float* dbi1 = (const float*)d_in[15];
    const float* dbh1 = (const float*)d_in[16];
    const float* fcW  = (const float*)d_in[17];
    const float* fcB  = (const float*)d_in[18];
    float* out = (float*)d_out;

    dim3 grid(B_TOT / BPB);
    lstm_enc_dec<<<grid, NT, 0, stream>>>(x,
                                          eW0, eU0, ebi0, ebh0,
                                          eW1, eU1, ebi1, ebh1,
                                          dW0, dU0, dbi0, dbh0,
                                          dW1, dU1, dbi1, dbh1,
                                          fcW, fcB, out);
}
